// MultiHeadAttention_4380866642082
// MI455X (gfx1250) — matrix operand order repaired and checked by reading, not compiled, not run on hardware
//
#include <hip/hip_runtime.h>
#include <hip/hip_bf16.h>

// ---------------------------------------------------------------------------
// MI455X (gfx1250) multi-head attention with RoPE.
// bf16 WMMA (v_wmma_f32_16x16x32_bf16) for all matrix math, TDM
// (tensor_load_to_lds + s_wait_tensorcnt) for row-major tile staging,
// global_prefetch_b8 for latency hiding.
// ---------------------------------------------------------------------------

#define D_MODEL   1024
#define NUM_HEADS 16
#define HEAD_DIM  64
#define BATCH     2
#define SEQ       2048
#define MTOT      (BATCH * SEQ)     // 4096 rows of (b,s)
#define NQKV      (3 * D_MODEL)     // 3072

typedef __attribute__((ext_vector_type(16))) __bf16        v16bf;
typedef __attribute__((ext_vector_type(8)))  float         v8f;
typedef __attribute__((ext_vector_type(4)))  unsigned int  u32x4;
typedef __attribute__((ext_vector_type(8)))  int           i32x8;
typedef __attribute__((ext_vector_type(4)))  int           i32x4;

// float -> bf16 bits, round-to-nearest-even
__device__ __forceinline__ unsigned short f2bf(float f) {
  unsigned int u = __builtin_bit_cast(unsigned int, f);
  unsigned int r = (u + 0x7FFFu + ((u >> 16) & 1u)) >> 16;
  return (unsigned short)r;
}

// ---- Tensor Data Mover: 2D tile (rows x cols, bf16) global -> LDS ---------
// D# per CDNA5 ISA. tile dims == tensor dims (in-bounds
// by construction). data_size=1 (2 bytes). Issued wave-uniform; EXEC ignored.
#if __has_builtin(__builtin_amdgcn_tensor_load_to_lds)
#define HAS_TDM 1
#else
#define HAS_TDM 0
#endif

__device__ __forceinline__ void tdm_load_tile_bf16(unsigned lds_off, const void* gptr,
                                                   unsigned cols, unsigned rows,
                                                   unsigned long long row_stride_elems) {
#if HAS_TDM
  // Group 0: count=1 | lds_addr[63:32] | global_addr[120:64] | type=2 [127:126]
  struct { unsigned long long a, b; } p;
  p.a = 1ull | ((unsigned long long)lds_off << 32);
  unsigned long long ga = (unsigned long long)(size_t)gptr;
  p.b = (ga & 0x01FFFFFFFFFFFFFFull) | (2ull << 62);
  u32x4 g0 = __builtin_bit_cast(u32x4, p);

  // Group 1: wg_mask=0, data_size=1(2B) at [17:16], flags=0,
  // tensor_dim0[79:48], tensor_dim1[111:80], tile_dim0[127:112],
  // tile_dim1[143:128], tile_dim2[159:144]=0, dim0_stride[207:160],
  // dim1_stride[255:208]=0 (2D).
  struct { unsigned long long q0, q1, q2, q3; } q;
  q.q0 = (1ull << 16) | ((unsigned long long)(cols & 0xFFFFu) << 48);
  q.q1 = ((unsigned long long)(cols >> 16) & 0xFFFFull)
       | ((unsigned long long)rows << 16)
       | ((unsigned long long)(cols & 0xFFFFu) << 48);
  q.q2 = (unsigned long long)(rows & 0xFFFFu)
       | ((row_stride_elems & 0xFFFFFFFFull) << 32);
  q.q3 = ((row_stride_elems >> 32) & 0xFFFFull);
  i32x8 g1 = __builtin_bit_cast(i32x8, q);

  i32x4 z4 = {0, 0, 0, 0};
#if defined(__clang_major__) && (__clang_major__ >= 23)
  i32x8 z8 = {0, 0, 0, 0, 0, 0, 0, 0};
  __builtin_amdgcn_tensor_load_to_lds(g0, g1, z4, z4, z8, 0);
#else
  __builtin_amdgcn_tensor_load_to_lds(g0, g1, z4, z4, 0);
#endif
#endif
}

__device__ __forceinline__ void tdm_wait() {
#if HAS_TDM
  __builtin_amdgcn_s_wait_tensorcnt(0);
#endif
}

// ---- WMMA fragment loaders (ISA 7.12.2 layouts) ---------------------------
// A 16x32 bf16 from row-major [m][k], stride ld (u16 units). lane<16: K in
// {0..7,16..23}; lane>=16: K in {8..15,24..31}; m = lane&15. Packed reads.
__device__ __forceinline__ v16bf load_a_frag(const unsigned short* base, int ld) {
  int lane = threadIdx.x & 31;
  int m = lane & 15, half = lane >> 4;
  v16bf a;
#pragma unroll
  for (int j = 0; j < 8; ++j) {
    int k = ((j < 4) ? (2 * j) : (2 * j + 8)) + 8 * half;
    unsigned int u = *(const unsigned int*)(base + m * ld + k);
    a[2 * j]     = __builtin_bit_cast(__bf16, (unsigned short)(u & 0xffffu));
    a[2 * j + 1] = __builtin_bit_cast(__bf16, (unsigned short)(u >> 16));
  }
  return a;
}

// B 32x16 bf16 from row-major [k][n], stride ld. n = lane&15; elements = K in A's order.
__device__ __forceinline__ v16bf load_b_frag(const unsigned short* base, int ld) {
  int lane = threadIdx.x & 31;
  int n = lane & 15, half = lane >> 4;
  v16bf b;
#pragma unroll
  for (int j = 0; j < 8; ++j) {
    int k = ((j < 4) ? (2 * j) : (2 * j + 8)) + 8 * half;
    b[2 * j]     = __builtin_bit_cast(__bf16, base[k * ld + n]);
    b[2 * j + 1] = __builtin_bit_cast(__bf16, base[(k + 1) * ld + n]);
  }
  return b;
}

// ---------------------------------------------------------------------------
// Kernel 1: QKV = x @ w_qkv with fused RoPE on Q/K.
// WG tile 128(M) x 128(N), BK=32, 256 threads = 8 waves (4 m-sub x 2 n-sub);
// each wave computes 32x64 (8 WMMAs per k-step). Wave's 64-col span = one head.
// Output: q/k/v bf16, layout [B*H][S][64].
// ---------------------------------------------------------------------------
__global__ void __launch_bounds__(256)
qkv_rope_kernel(const float* __restrict__ x, const float* __restrict__ w,
                unsigned short* __restrict__ qws, unsigned short* __restrict__ kws,
                unsigned short* __restrict__ vws) {
  __shared__ unsigned short Xs[128 * 32];   // 8 KB
  __shared__ unsigned short Wsh[32 * 128];  // 8 KB

  int tid = threadIdx.x;
  int wid = tid >> 5, lane = tid & 31;
  int msub = wid & 3, nsub = wid >> 2;
  int bm = blockIdx.x * 128;
  int bn = blockIdx.y * 128;
  int n = lane & 15, half = lane >> 4;

  v8f acc[2][4] = {{v8f{}, v8f{}, v8f{}, v8f{}}, {v8f{}, v8f{}, v8f{}, v8f{}}};

  for (int kk = 0; kk < D_MODEL; kk += 32) {
    __syncthreads();
#pragma unroll
    for (int i = 0; i < 16; ++i) {          // X tile 128x32 f32 -> bf16
      int idx = tid + 256 * i;
      int r = idx >> 5, c = idx & 31;
      Xs[idx] = f2bf(x[(size_t)(bm + r) * D_MODEL + kk + c]);
      if (kk + 32 < D_MODEL && i == 0)
        __builtin_prefetch(&x[(size_t)(bm + r) * D_MODEL + kk + 32 + c], 0, 1);
    }
#pragma unroll
    for (int i = 0; i < 16; ++i) {          // W tile 32x128 f32 -> bf16
      int idx = tid + 256 * i;
      int r = idx >> 7, c = idx & 127;
      Wsh[idx] = f2bf(w[(size_t)(kk + r) * NQKV + bn + c]);
      if (kk + 32 < D_MODEL && i < 4)
        __builtin_prefetch(&w[(size_t)(kk + 32 + r + i * 8) * NQKV + bn + c], 0, 1);
    }
    __syncthreads();

    v16bf a0 = load_a_frag(&Xs[(msub * 32) * 32], 32);
    v16bf a1 = load_a_frag(&Xs[(msub * 32 + 16) * 32], 32);
#pragma unroll
    for (int t = 0; t < 4; ++t) {
      v16bf b = load_b_frag(&Wsh[nsub * 64 + t * 16], 128);
      acc[0][t] = __builtin_amdgcn_wmma_f32_16x16x32_bf16(false, a0, false, b,
                                                          (short)0, acc[0][t], false, false);
      acc[1][t] = __builtin_amdgcn_wmma_f32_16x16x32_bf16(false, a1, false, b,
                                                          (short)0, acc[1][t], false, false);
    }
  }

  // Epilogue: wave covers 64 aligned columns -> exactly one (q/k/v, head).
  int gc0 = bn + nsub * 64;
  int which = gc0 >> 10;                    // 0=q, 1=k, 2=v
  int h = (gc0 & 1023) >> 6;
  unsigned short* dst = (which == 0) ? qws : ((which == 1) ? kws : vws);

#pragma unroll
  for (int mt = 0; mt < 2; ++mt) {
#pragma unroll
    for (int v = 0; v < 8; ++v) {
      int row = bm + msub * 32 + mt * 16 + v + 8 * half;  // global (b,s) row
      int b = row >> 11, s = row & (SEQ - 1);
      unsigned short* rowp = dst + (((size_t)(b * NUM_HEADS + h) * SEQ + s) << 6);
      if (which < 2) {
        // RoPE: pair (d, d+32) lives in tiles (t, t+2) at same lane/vgpr.
#pragma unroll
        for (int t = 0; t < 2; ++t) {
          int i = t * 16 + n;               // 0..31 freq index
          float x1 = acc[mt][t][v];
          float x2 = acc[mt][t + 2][v];
          float ang = (float)s * __expf((float)i * -0.28782313662425572f); // ln(1e4)/32
          float sn, cs;
          __sincosf(ang, &sn, &cs);
          rowp[i]      = f2bf(x1 * cs - x2 * sn);
          rowp[i + 32] = f2bf(x2 * cs + x1 * sn);
        }
      } else {
#pragma unroll
        for (int t = 0; t < 4; ++t)
          rowp[t * 16 + n] = f2bf(acc[mt][t][v]);
      }
    }
  }
}

// ---------------------------------------------------------------------------
// Kernel 2: flash-style causal attention. One (b,h) per blockIdx.y, 64 query
// rows per WG (4 waves x 16 rows). V tile staged by the TDM (row-major fit);
// K staged manually transposed (TDM cannot transpose) so QK^T B-fragments are
// row-major [d][key] reads. P round-trips per-wave LDS (C-layout -> A-layout).
// ---------------------------------------------------------------------------
__global__ void __launch_bounds__(128)
attn_kernel(const unsigned short* __restrict__ qws, const unsigned short* __restrict__ kws,
            const unsigned short* __restrict__ vws, unsigned short* __restrict__ aows) {
  __shared__ unsigned short Kt[64 * 32];      // 4 KB: K^T tile [d][key]
  __shared__ unsigned short Vs[32 * 64];      // 4 KB: V tile  [key][d]
  __shared__ unsigned short Ps[4][16 * 32];   // 4 KB: per-wave P scratch [m][k]

  int tid = threadIdx.x;                      // 0..127
  int wid = tid >> 5, lane = tid & 31;
  int n = lane & 15, half = lane >> 4;
  int qb = blockIdx.x;                        // query block
  int bh = blockIdx.y;                        // b*H + h
  int q0 = qb * 64 + wid * 16;

  const unsigned short* Qbase = qws + ((size_t)bh * SEQ + q0) * HEAD_DIM;
  v16bf aq0 = load_a_frag(Qbase, HEAD_DIM);         // hd 0..31
  v16bf aq1 = load_a_frag(Qbase + 32, HEAD_DIM);    // hd 32..63

  v8f o[4] = {v8f{}, v8f{}, v8f{}, v8f{}};
  float rmax[8], rsum[8];
#pragma unroll
  for (int v = 0; v < 8; ++v) { rmax[v] = -1e30f; rsum[v] = 0.f; }

  const unsigned short* Kbh = kws + (size_t)bh * SEQ * HEAD_DIM;
  const unsigned short* Vbh = vws + (size_t)bh * SEQ * HEAD_DIM;
  int nkb = 2 * qb + 2;                       // causal: keys up to qb*64+63

  for (int kb = 0; kb < nkb; ++kb) {
    int k0 = kb * 32;
    __syncthreads();

    // V tile via Tensor Data Mover (wave 0 issues; EXEC-independent DMA).
    if (HAS_TDM) {
      if (wid == 0)
        tdm_load_tile_bf16((unsigned)(size_t)&Vs[0], &Vbh[(size_t)k0 * HEAD_DIM],
                           /*cols=*/HEAD_DIM, /*rows=*/32, /*stride=*/HEAD_DIM);
    } else {
#pragma unroll
      for (int i = 0; i < 8; ++i) {
        int idx = tid + 128 * i;
        int key = idx >> 5, dp = idx & 31;
        *(unsigned int*)&Vs[key * 64 + dp * 2] =
            *(const unsigned int*)&Vbh[(size_t)(k0 + key) * HEAD_DIM + dp * 2];
      }
    }

    // K^T staging (transpose): Kt[d][key] <- K[key][d].
#pragma unroll
    for (int i = 0; i < 16; ++i) {
      int idx = tid + 128 * i;                // 0..2047
      int key = idx >> 6, d = idx & 63;
      Kt[d * 32 + key] = Kbh[(size_t)(k0 + key) * HEAD_DIM + d];
    }

    // Prefetch next K/V block into cache while we compute this one.
    if (kb + 1 < nkb) {
      __builtin_prefetch(&Kbh[(size_t)(k0 + 32) * HEAD_DIM + tid * 16], 0, 1);
      __builtin_prefetch(&Vbh[(size_t)(k0 + 32) * HEAD_DIM + tid * 16], 0, 1);
    }

    if (wid == 0) tdm_wait();                 // TENSORcnt==0 before barrier
    __syncthreads();

    // S(16x32) = Q(16x64) * K^T(64x32): two 16x16 tiles, 2 WMMAs each.
    v8f sc[2];
#pragma unroll
    for (int t = 0; t < 2; ++t) {
      v8f z = v8f{};
      v16bf b0 = load_b_frag(&Kt[t * 16], 32);            // hd rows 0..31
      v16bf b1 = load_b_frag(&Kt[32 * 32 + t * 16], 32);  // hd rows 32..63
      z = __builtin_amdgcn_wmma_f32_16x16x32_bf16(false, aq0, false, b0, (short)0, z, false, false);
      z = __builtin_amdgcn_wmma_f32_16x16x32_bf16(false, aq1, false, b1, (short)0, z, false, false);
      sc[t] = z;
    }

    // scale + causal mask + online softmax (rows owned per lane-half).
    float bmax[8];
#pragma unroll
    for (int v = 0; v < 8; ++v) {
      int m = v + 8 * half;
      int qg = q0 + m;
#pragma unroll
      for (int t = 0; t < 2; ++t) {
        int kg = k0 + t * 16 + n;
        float s = sc[t][v] * 0.125f;          // 1/sqrt(64)
        if (kg > qg) s = -1e30f;
        sc[t][v] = s;
      }
      float mx = fmaxf(sc[0][v], sc[1][v]);
#pragma unroll
      for (int off = 1; off < 16; off <<= 1)  // reduce within 16-lane half
        mx = fmaxf(mx, __shfl_xor(mx, off, 32));
      bmax[v] = mx;
    }
#pragma unroll
    for (int v = 0; v < 8; ++v) {
      float nm = fmaxf(rmax[v], bmax[v]);
      float alpha = __expf(rmax[v] - nm);
      rmax[v] = nm;
      float p0 = __expf(sc[0][v] - nm);
      float p1 = __expf(sc[1][v] - nm);
      float rs = p0 + p1;
#pragma unroll
      for (int off = 1; off < 16; off <<= 1)
        rs += __shfl_xor(rs, off, 32);
      rsum[v] = rsum[v] * alpha + rs;
#pragma unroll
      for (int t = 0; t < 4; ++t) o[t][v] *= alpha;
      int m = v + 8 * half;                   // C-layout -> LDS row-major P
      Ps[wid][m * 32 + n]      = f2bf(p0);
      Ps[wid][m * 32 + 16 + n] = f2bf(p1);
    }

    // O(16x64) += P(16x32) * V(32x64): single A frag, 4 WMMAs.
    v16bf ap = load_a_frag(&Ps[wid][0], 32);
#pragma unroll
    for (int t = 0; t < 4; ++t) {
      v16bf bv = load_b_frag(&Vs[t * 16], 64);
      o[t] = __builtin_amdgcn_wmma_f32_16x16x32_bf16(false, ap, false, bv,
                                                     (short)0, o[t], false, false);
    }
  }

  // Normalize and write attn output in [B*S][D_MODEL] bf16 for the out-proj.
  int b = bh >> 4, h = bh & 15;
#pragma unroll
  for (int v = 0; v < 8; ++v) {
    int m = v + 8 * half;
    size_t row = (size_t)b * SEQ + q0 + m;
    float inv = 1.0f / rsum[v];
#pragma unroll
    for (int t = 0; t < 4; ++t)
      aows[row * D_MODEL + h * 64 + t * 16 + n] = f2bf(o[t][v] * inv);
  }
}

// ---------------------------------------------------------------------------
// Kernel 3: out = attn @ w_out, fp32 output. WG tile 128x128, BK=32.
// bf16 A tile staged by TDM; W tile converted f32->bf16 manually.
// ---------------------------------------------------------------------------
__global__ void __launch_bounds__(256)
out_proj_kernel(const unsigned short* __restrict__ a_bf, const float* __restrict__ w,
                float* __restrict__ out) {
  __shared__ unsigned short As[128 * 32];   // 8 KB
  __shared__ unsigned short Wsh[32 * 128];  // 8 KB

  int tid = threadIdx.x;
  int wid = tid >> 5, lane = tid & 31;
  int msub = wid & 3, nsub = wid >> 2;
  int bm = blockIdx.x * 128;
  int bn = blockIdx.y * 128;
  int n = lane & 15, half = lane >> 4;

  v8f acc[2][4] = {{v8f{}, v8f{}, v8f{}, v8f{}}, {v8f{}, v8f{}, v8f{}, v8f{}}};

  for (int kk = 0; kk < D_MODEL; kk += 32) {
    __syncthreads();

    if (HAS_TDM) {
      if (wid == 0)
        tdm_load_tile_bf16((unsigned)(size_t)&As[0], &a_bf[(size_t)bm * D_MODEL + kk],
                           /*cols=*/32, /*rows=*/128, /*stride=*/D_MODEL);
    } else {
#pragma unroll
      for (int i = 0; i < 8; ++i) {
        int idx = tid + 256 * i;              // uints
        int r = idx >> 4, cp = idx & 15;
        *(unsigned int*)&As[r * 32 + cp * 2] =
            *(const unsigned int*)&a_bf[(size_t)(bm + r) * D_MODEL + kk + cp * 2];
      }
    }
#pragma unroll
    for (int i = 0; i < 16; ++i) {            // W tile 32x128 f32 -> bf16
      int idx = tid + 256 * i;
      int r = idx >> 7, c = idx & 127;
      Wsh[idx] = f2bf(w[(size_t)(kk + r) * D_MODEL + bn + c]);
      if (kk + 32 < D_MODEL && i < 4)
        __builtin_prefetch(&w[(size_t)(kk + 32 + r + i * 8) * D_MODEL + bn + c], 0, 1);
    }
    if (wid == 0) tdm_wait();
    __syncthreads();

    v16bf a0 = load_a_frag(&As[(msub * 32) * 32], 32);
    v16bf a1 = load_a_frag(&As[(msub * 32 + 16) * 32], 32);
#pragma unroll
    for (int t = 0; t < 4; ++t) {
      v16bf b = load_b_frag(&Wsh[nsub * 64 + t * 16], 128);
      acc[0][t] = __builtin_amdgcn_wmma_f32_16x16x32_bf16(false, a0, false, b,
                                                          (short)0, acc[0][t], false, false);
      acc[1][t] = __builtin_amdgcn_wmma_f32_16x16x32_bf16(false, a1, false, b,
                                                          (short)0, acc[1][t], false, false);
    }
  }

#pragma unroll
  for (int mt = 0; mt < 2; ++mt) {
#pragma unroll
    for (int v = 0; v < 8; ++v) {
      size_t row = (size_t)(bm + msub * 32 + mt * 16 + v + 8 * half);
#pragma unroll
      for (int t = 0; t < 4; ++t) {
        int col = bn + nsub * 64 + t * 16 + n;
        out[row * D_MODEL + col] = acc[mt][t][v];
      }
    }
  }
}

// ---------------------------------------------------------------------------
// Launch: inputs = {x, mask(causal, computed analytically), w_qkv, w_out}.
// Workspace (bf16): Q | K | V (each [B*H][S][64]) | attn_out [B*S][D] = 32 MB.
// ---------------------------------------------------------------------------
extern "C" void kernel_launch(void* const* d_in, const int* in_sizes, int n_in,
                              void* d_out, int out_size, void* d_ws, size_t ws_size,
                              hipStream_t stream) {
  const float* x     = (const float*)d_in[0];
  const float* w_qkv = (const float*)d_in[2];
  const float* w_out = (const float*)d_in[3];
  float* out = (float*)d_out;

  size_t seg = (size_t)BATCH * NUM_HEADS * SEQ * HEAD_DIM;  // 4,194,304 elems
  unsigned short* qws  = (unsigned short*)d_ws;
  unsigned short* kws  = qws + seg;
  unsigned short* vws  = kws + seg;
  unsigned short* aows = vws + seg;

  qkv_rope_kernel<<<dim3(MTOT / 128, NQKV / 128), 256, 0, stream>>>(x, w_qkv, qws, kws, vws);
  attn_kernel<<<dim3(SEQ / 64, BATCH * NUM_HEADS), 128, 0, stream>>>(qws, kws, vws, aows);
  out_proj_kernel<<<dim3(MTOT / 128, D_MODEL / 128), 256, 0, stream>>>(aows, w_out, out);
}
